// LSTMStrideModel_36361193128225
// MI455X (gfx1250) — matrix operand order repaired and checked by reading, not compiled, not run on hardware
//
#include <hip/hip_runtime.h>

// ---------------------------------------------------------------------------
// Fused LSTM (B=8192, T=128, I=32, H=64) + final FC (8 out), CDNA5 WMMA.
// One wave32 owns a 16-row batch tile for the whole T loop.
// Per timestep: gates[16,256] = xA(16x32)*Bih + hA(16x64)*Bhh + bias
//   -> 48 x v_wmma_f32_16x16x32_f16 per wave per timestep.
// Weights pre-swizzled into B-fragment layout in LDS (2 x ds_load_b128/frag).
// Fragment loads for each jt group are staged before the WMMAs so the DS
// waits are partial (pipelined) instead of full drains.
// Accumulators start from inline-0 C; bias folded into the elementwise phase.
// Activations use native v_rcp_f32/v_exp_f32 (no precise-division expansion).
// An opaque per-iteration zero keeps the fragment loads inside the t-loop
// (otherwise LICM hoists 48 frags = 384 VGPRs and spills them to scratch).
// h kept in per-wave LDS (f16, plain [m][j]); c kept in VGPRs (C-layout).
// ---------------------------------------------------------------------------

typedef __attribute__((ext_vector_type(16))) _Float16 v16h;
typedef __attribute__((ext_vector_type(8)))  _Float16 v8h;
typedef __attribute__((ext_vector_type(2)))  __fp16   fp16x2;
typedef __attribute__((ext_vector_type(8)))  float    v8f;
typedef __attribute__((ext_vector_type(4)))  float    v4f;

#define INPUT_SZ   32
#define HIDDEN     64
#define T_STEPS    128
#define BATCH      8192
#define OUT_SZ     8
#define WAVES_PB   8
#define ROWS_PW    16
#define BLOCK_THREADS (WAVES_PB * 32)

static __device__ __forceinline__ v16h catv(v8h lo, v8h hi) {
  return __builtin_shufflevector(lo, hi, 0,1,2,3,4,5,6,7,8,9,10,11,12,13,14,15);
}

static __device__ __forceinline__ v8h cvt8(v4f a, v4f b) {
  fp16x2 p0 = __builtin_amdgcn_cvt_pkrtz(a[0], a[1]);
  fp16x2 p1 = __builtin_amdgcn_cvt_pkrtz(a[2], a[3]);
  fp16x2 p2 = __builtin_amdgcn_cvt_pkrtz(b[0], b[1]);
  fp16x2 p3 = __builtin_amdgcn_cvt_pkrtz(b[2], b[3]);
  v8h r = { (_Float16)p0[0], (_Float16)p0[1], (_Float16)p1[0], (_Float16)p1[1],
            (_Float16)p2[0], (_Float16)p2[1], (_Float16)p3[0], (_Float16)p3[1] };
  return r;
}

// B fragment: lane holds 16 f16 (two 16B chunks) pre-swizzled in LDS.
static __device__ __forceinline__ v16h loadB(const _Float16* frag, int lane) {
  const v8h* p = (const v8h*)(frag + lane * 16);
  return catv(p[0], p[1]);
}

// A fragment (16x32 f16) from plain [m][64] f16 LDS buffer, column offset j0.
static __device__ __forceinline__ v16h loadA(const _Float16* hw, int m, int half, int j0) {
  const v8h* lo = (const v8h*)(hw + m * HIDDEN + j0 + half * 8);
  const v8h* hi = (const v8h*)(hw + m * HIDDEN + j0 + 16 + half * 8);
  return catv(*lo, *hi);
}

static __device__ __forceinline__ v8f wmma_f16(v16h a, v16h b, v8f c) {
  return __builtin_amdgcn_wmma_f32_16x16x32_f16(
      false, a, false, b, (short)0, c, false, false);
}

static __device__ __forceinline__ v8f splat8(float x) {
  v8f r = { x, x, x, x, x, x, x, x };
  return r;
}

// Native-instruction activations: v_exp_f32 + v_rcp_f32, no div expansion.
static __device__ __forceinline__ float fsigm(float x) {
  return __builtin_amdgcn_rcpf(1.0f + __expf(-x));           // 1/(1+e^-x)
}
static __device__ __forceinline__ float ftanh(float x) {
  // tanh(x) = 1 - 2/(e^{2x}+1)
  return __builtin_fmaf(-2.0f,
                        __builtin_amdgcn_rcpf(__expf(2.0f * x) + 1.0f),
                        1.0f);
}

__global__ __launch_bounds__(BLOCK_THREADS, 4)
void lstm_wmma_kernel(const float* __restrict__ x,
                      const float* __restrict__ W_ih,
                      const float* __restrict__ W_hh,
                      const float* __restrict__ b_ih,
                      const float* __restrict__ b_hh,
                      const float* __restrict__ W_fc,
                      const float* __restrict__ b_fc,
                      float* __restrict__ out) {
  // --- LDS: pre-swizzled f16 B-fragments + per-wave h buffers -------------
  __shared__ __align__(16) _Float16 sBih[16 * 256];   //  8 KB: 16 N-tiles (K=32)
  __shared__ __align__(16) _Float16 sBhh[32 * 256];   // 16 KB: 2 K-halves x 16 N-tiles
  __shared__ __align__(16) _Float16 sH[WAVES_PB][ROWS_PW * HIDDEN]; // 16 KB

  // ---- one-time weight swizzle (f32 -> f16, B-fragment VGPR order) -------
  // fragment elem (lane l, f16 slot q): n = nb*16 + (l&15), k = kbase + (l>>4)*8 + (q&7) + (q>>3)*16
  for (int i = threadIdx.x; i < 16 * 256; i += BLOCK_THREADS) {
    int f = i >> 8;          // N tile
    int w = i & 255;
    int l = w >> 4, q = w & 15;
    int n = f * 16 + (l & 15);
    int k = (l >> 4) * 8 + (q & 7) + (q >> 3) * 16;           // K = 32
    sBih[i] = (_Float16)W_ih[n * INPUT_SZ + k];
  }
  for (int i = threadIdx.x; i < 32 * 256; i += BLOCK_THREADS) {
    int f = i >> 8;
    int kb = f >> 4, nb = f & 15;
    int w = i & 255;
    int l = w >> 4, q = w & 15;
    int n = nb * 16 + (l & 15);
    int k = kb * 32 + (l >> 4) * 8 + (q & 7) + (q >> 3) * 16; // K = 64 split in two 32-chunks
    sBhh[i] = (_Float16)W_hh[n * HIDDEN + k];
  }
  __syncthreads();

  // ---- per-wave setup ----------------------------------------------------
  const int lane = threadIdx.x & 31;
  const int wave = threadIdx.x >> 5;
  const int m    = lane & 15;   // batch row within tile / N column index
  const int half = lane >> 4;
  const int rowBase = (blockIdx.x * WAVES_PB + wave) * ROWS_PW;

  _Float16* hw = &sH[wave][0];
  for (int i = lane; i < ROWS_PW * HIDDEN; i += 32) hw[i] = (_Float16)0.0f;

  // bias per lane depends only on N = gate_tile*16 + (lane&15); kept as
  // 16 scalar floats and applied in the elementwise phase (not splatted).
  float biasv[16];
#pragma unroll
  for (int g = 0; g < 16; ++g) {
    int n = g * 16 + m;
    biasv[g] = b_ih[n] + b_hh[n];
  }

  // per-lane x pointer: row = rowBase + m, K-offset half*8
  const float* xlane = x + ((size_t)(rowBase + m) * T_STEPS) * INPUT_SZ + half * 8;

  v8f cs0 = splat8(0.0f), cs1 = splat8(0.0f);
  v8f cs2 = splat8(0.0f), cs3 = splat8(0.0f);   // cell state, C-layout

#pragma unroll 1
  for (int t = 0; t < T_STEPS; ++t) {
    // Opaque zero: defeats LICM so B-fragments are ds_load'ed every
    // iteration instead of being hoisted (384 VGPRs) and spilled to scratch.
    int zero;
    asm volatile("s_mov_b32 %0, 0" : "=s"(zero));
    const _Float16* bih = sBih + zero;
    const _Float16* bhh = sBhh + zero;

    // A fragments of previous h (16x64 -> two 16x32 frags)
    v16h hA0 = loadA(hw, m, half, 0);
    v16h hA1 = loadA(hw, m, half, 32);

    // A fragment of x_t (f32 global -> f16), 4 x 16B coalesced loads
    const float* xt = xlane + t * INPUT_SZ;
    v4f x0 = *(const v4f*)(xt);
    v4f x1 = *(const v4f*)(xt + 4);
    v4f x2 = *(const v4f*)(xt + 16);
    v4f x3 = *(const v4f*)(xt + 20);
    v16h xA = catv(cvt8(x0, x1), cvt8(x2, x3));

#pragma unroll
    for (int jt = 0; jt < 4; ++jt) {
      // ---- stage all 12 B fragments for this gate group up front ----------
      v16h xB[4], h0B[4], h1B[4];
#pragma unroll
      for (int g = 0; g < 4; ++g) {
        xB[g]  = loadB(&bih[(g * 4 + jt) * 256], lane);
        h0B[g] = loadB(&bhh[(g * 4 + jt) * 256], lane);
        h1B[g] = loadB(&bhh[(16 + g * 4 + jt) * 256], lane);
      }

      v8f zacc = splat8(0.0f);
      v8f ai = wmma_f16(xA, xB[0], zacc);
      v8f af = wmma_f16(xA, xB[1], zacc);
      v8f ag = wmma_f16(xA, xB[2], zacc);
      v8f ao = wmma_f16(xA, xB[3], zacc);

      ai = wmma_f16(hA0, h0B[0], ai);
      af = wmma_f16(hA0, h0B[1], af);
      ag = wmma_f16(hA0, h0B[2], ag);
      ao = wmma_f16(hA0, h0B[3], ao);

      ai = wmma_f16(hA1, h1B[0], ai);
      af = wmma_f16(hA1, h1B[1], af);
      ag = wmma_f16(hA1, h1B[2], ag);
      ao = wmma_f16(hA1, h1B[3], ao);

      const float bi = biasv[jt];
      const float bf = biasv[4 + jt];
      const float bg = biasv[8 + jt];
      const float bo = biasv[12 + jt];

      v8f* cs = (jt == 0) ? &cs0 : (jt == 1) ? &cs1 : (jt == 2) ? &cs2 : &cs3;
#pragma unroll
      for (int v = 0; v < 8; ++v) {
        float iv = fsigm(ai[v] + bi);
        float fv = fsigm(af[v] + bf);
        float gv = ftanh(ag[v] + bg);
        float ov = fsigm(ao[v] + bo);
        float cc = fv * (*cs)[v] + iv * gv;
        (*cs)[v] = cc;
        float hh = ov * ftanh(cc);
        // C-layout element (M = half*8+v, N = jt*16+m) -> plain [m][j] buffer
        hw[(half * 8 + v) * HIDDEN + jt * 16 + m] = (_Float16)hh;
      }
    }
  }

  // ---- tail FC: out[row, :8] = h_last @ W_fc^T + b_fc --------------------
  if (lane < 16) {
    int row = rowBase + m;
#pragma unroll
    for (int o = 0; o < OUT_SZ; ++o) {
      float acc = b_fc[o];
#pragma unroll 8
      for (int j = 0; j < HIDDEN; ++j)
        acc += (float)hw[m * HIDDEN + j] * W_fc[o * HIDDEN + j];
      out[row * OUT_SZ + o] = acc;
    }
  }
}

extern "C" void kernel_launch(void* const* d_in, const int* in_sizes, int n_in,
                              void* d_out, int out_size, void* d_ws, size_t ws_size,
                              hipStream_t stream) {
  (void)in_sizes; (void)n_in; (void)out_size; (void)d_ws; (void)ws_size;
  const float* x    = (const float*)d_in[0];
  const float* W_ih = (const float*)d_in[1];
  const float* W_hh = (const float*)d_in[2];
  const float* b_ih = (const float*)d_in[3];
  const float* b_hh = (const float*)d_in[4];
  const float* W_fc = (const float*)d_in[5];
  const float* b_fc = (const float*)d_in[6];
  float* out = (float*)d_out;

  dim3 grid(BATCH / (WAVES_PB * ROWS_PW));  // 64 blocks
  dim3 block(BLOCK_THREADS);                // 256 threads = 8 waves
  lstm_wmma_kernel<<<grid, block, 0, stream>>>(x, W_ih, W_hh, b_ih, b_hh,
                                               W_fc, b_fc, out);
}
